// SA_Block_69114613727431
// MI455X (gfx1250) — compile-verified, matrix operand order checked by reading, not run on hardware
//
#include <hip/hip_runtime.h>

typedef __attribute__((ext_vector_type(16))) _Float16 v16h;
typedef __attribute__((ext_vector_type(8)))  _Float16 v8h;
typedef __attribute__((ext_vector_type(8)))  float    v8f;

#define B_  4
#define C_  256
#define CQ_ 32
#define N_  4096

// Load a 16x32 (rows x K) f16 tile stored row-major with given row stride
// into the CDNA5 WMMA 16-bit A/B register striping:
//   lanes 0-15 : row = lane,    K = 0..7 (VGPR0-3) and 16..23 (VGPR4-7)
//   lanes 16-31: row = lane-16, K = 8..15           and 24..31
// i.e. two contiguous 16-byte chunks per lane.
__device__ __forceinline__ v16h load_tile16x32(const _Float16* base, int strideHalfs) {
  const int l  = threadIdx.x & 31;
  const int hf = l >> 4;
  const _Float16* p = base + (size_t)(l & 15) * strideHalfs + hf * 8;
  union { v16h v; v8h h[2]; } u;
  u.h[0] = *(const v8h*)(p);
  u.h[1] = *(const v8h*)(p + 16);
  return u.v;
}

// ---------------- kernel 1: pack Wq/Wk/Wv into one 320x256 f16 matrix ----------
__global__ void pack_weights(const float* __restrict__ Wq, const float* __restrict__ bq,
                             const float* __restrict__ Wk, const float* __restrict__ bk,
                             const float* __restrict__ Wv, const float* __restrict__ bv,
                             _Float16* __restrict__ Wh, float* __restrict__ bP) {
  int idx = blockIdx.x * blockDim.x + threadIdx.x;
  if (idx >= 320 * 256) return;
  int o = idx >> 8, c = idx & 255;
  float w = (o < 32) ? Wq[o * 256 + c] : (o < 64) ? Wk[(o - 32) * 256 + c]
                                                  : Wv[(o - 64) * 256 + c];
  Wh[idx] = (_Float16)w;
  if (c == 0) bP[o] = (o < 32) ? bq[o] : (o < 64) ? bk[o - 32] : bv[o - 64];
}

// ------------- kernel 2: x [B,C,N] f32 -> xT [B,N,C] f16 (LDS tile transpose) --
__global__ __launch_bounds__(256) void transpose_cvt(const float* __restrict__ x,
                                                     _Float16* __restrict__ xt) {
  __shared__ _Float16 t[32][33];
  const int b = blockIdx.z, ct = blockIdx.y, nt = blockIdx.x;
  const int lx = threadIdx.x & 31;
  const int ly = threadIdx.x >> 5;   // 8 rows per pass
#pragma unroll
  for (int k = 0; k < 4; ++k) {
    int c = ct * 32 + ly + k * 8;
    int n = nt * 32 + lx;
    t[ly + k * 8][lx] = (_Float16)x[((size_t)b * C_ + c) * N_ + n];
  }
  __syncthreads();
#pragma unroll
  for (int k = 0; k < 4; ++k) {
    int n = nt * 32 + ly + k * 8;
    int c = ct * 32 + lx;
    xt[((size_t)b * N_ + n) * C_ + c] = t[lx][ly + k * 8];
  }
}

// ------------- kernel 3: fused QKV projection (one wave per 16x16 tile) --------
__global__ __launch_bounds__(32) void proj_kernel(
    const _Float16* __restrict__ Wh, const float* __restrict__ bP,
    const _Float16* __restrict__ xt, const float* __restrict__ mask,
    _Float16* __restrict__ Qh, _Float16* __restrict__ Kh, _Float16* __restrict__ Vh) {
  const int b = blockIdx.z, ot = blockIdx.y, nt = blockIdx.x;
  v8f acc = {0.f, 0.f, 0.f, 0.f, 0.f, 0.f, 0.f, 0.f};
#pragma unroll
  for (int ks = 0; ks < 8; ++ks) {
    v16h a  = load_tile16x32(Wh + (size_t)ot * 16 * C_ + ks * 32, C_);
    v16h bm = load_tile16x32(xt + ((size_t)b * N_ + nt * 16) * C_ + ks * 32, C_);
    acc = __builtin_amdgcn_wmma_f32_16x16x32_f16(false, a, false, bm, (short)0, acc,
                                                 false, false);
  }
  const int lane = threadIdx.x & 31;
  const int hf = lane >> 4, col = lane & 15;
  const int n = nt * 16 + col;
  const float mval = mask[(size_t)b * N_ + n];
#pragma unroll
  for (int r = 0; r < 8; ++r) {
    int o = ot * 16 + r + 8 * hf;
    float v = acc[r] + bP[o];
    if (o < 64) v *= mval;                        // mask applies to q and k only
    _Float16 h = (_Float16)v;
    if (o < 32)       Qh[((size_t)b * N_ + n) * CQ_ + o] = h;
    else if (o < 64)  Kh[((size_t)b * N_ + n) * CQ_ + (o - 32)] = h;
    else              Vh[((size_t)b * C_ + (o - 64)) * N_ + n] = h;
  }
}

// ------------- kernel 4: flash attention + gamma*out + x residual --------------
__global__ __launch_bounds__(64) void attn_kernel(
    const _Float16* __restrict__ Qh, const _Float16* __restrict__ Kh,
    const _Float16* __restrict__ Vh, const float* __restrict__ x,
    const float* __restrict__ gamma, float* __restrict__ out) {
  __shared__ __align__(16) _Float16 Plds[2][16 * 32];
  const int b    = blockIdx.y;
  const int i0   = blockIdx.x * 16;
  const int wave = threadIdx.x >> 5;
  const int lane = threadIdx.x & 31;
  const int hf   = lane >> 4, col = lane & 15;

  // Q tile (16 queries x 32 channels) kept in registers for the whole pass
  const v16h qa = load_tile16x32(Qh + ((size_t)b * N_ + i0) * CQ_, CQ_);

  const v8f vzero = {0.f, 0.f, 0.f, 0.f, 0.f, 0.f, 0.f, 0.f};
  v8f acc[8];
  float mrun[8], lrun[8];
#pragma unroll
  for (int t = 0; t < 8; ++t) acc[t] = vzero;
#pragma unroll
  for (int r = 0; r < 8; ++r) { mrun[r] = -3.0e38f; lrun[r] = 0.0f; }

  _Float16* pbuf = &Plds[wave][0];

  for (int j0 = 0; j0 < N_; j0 += 32) {
    // ---- energy tile: D(16x16) = Q^T(16x32) x K(32x16), K-dim == CQ == 32 ----
    const v16h kb0 = load_tile16x32(Kh + ((size_t)b * N_ + j0) * CQ_, CQ_);
    const v16h kb1 = load_tile16x32(Kh + ((size_t)b * N_ + j0 + 16) * CQ_, CQ_);
    v8f e0 = __builtin_amdgcn_wmma_f32_16x16x32_f16(false, qa, false, kb0, (short)0,
                                                    vzero, false, false);
    v8f e1 = __builtin_amdgcn_wmma_f32_16x16x32_f16(false, qa, false, kb1, (short)0,
                                                    vzero, false, false);

    if (j0 + 32 < N_) {
      __builtin_prefetch(Kh + ((size_t)b * N_ + j0 + 32) * CQ_ + lane * 2, 0, 0);
    }

    // ---- online softmax (row = r + 8*hf, cols live across the 16-lane half) --
    float corr[8];
#pragma unroll
    for (int r = 0; r < 8; ++r) {
      float cm = fmaxf(e0[r], e1[r]);
#pragma unroll
      for (int s = 1; s < 16; s <<= 1) cm = fmaxf(cm, __shfl_xor(cm, s, 32));
      const float mn = fmaxf(mrun[r], cm);
      const float cf = __expf(mrun[r] - mn);
      const float p0 = __expf(e0[r] - mn);
      const float p1 = __expf(e1[r] - mn);
      float ps = p0 + p1;
#pragma unroll
      for (int s = 1; s < 16; s <<= 1) ps += __shfl_xor(ps, s, 32);
      lrun[r] = lrun[r] * cf + ps;
      mrun[r] = mn;
      corr[r] = cf;
      const int row = r + 8 * hf;          // D-layout -> row-major P in LDS
      pbuf[row * 32 + col]      = (_Float16)p0;
      pbuf[row * 32 + 16 + col] = (_Float16)p1;
    }
    // DS ops are in-order within a wave; make the dependency explicit anyway.
    asm volatile("s_wait_dscnt 0" ::: "memory");
    const v16h pa = load_tile16x32(pbuf, 32);  // P as A-matrix (16x32)

    // ---- O += P(16x32) x V(32x16) per 16-channel tile; wave owns 128 chans ---
#pragma unroll
    for (int t = 0; t < 8; ++t) {
      const int c0 = wave * 128 + t * 16;
      const v16h vb = load_tile16x32(Vh + ((size_t)b * C_ + c0) * N_ + j0, N_);
#pragma unroll
      for (int r = 0; r < 8; ++r) acc[t][r] *= corr[r];
      acc[t] = __builtin_amdgcn_wmma_f32_16x16x32_f16(false, pa, false, vb, (short)0,
                                                      acc[t], false, false);
    }
  }

  // ---- epilogue: normalize, gamma * out + x ----
  const float g = gamma[0];
#pragma unroll
  for (int t = 0; t < 8; ++t) {
    const int c = wave * 128 + t * 16 + col;
#pragma unroll
    for (int r = 0; r < 8; ++r) {
      const int i = i0 + r + 8 * hf;
      const size_t off = ((size_t)b * C_ + c) * N_ + i;
      out[off] = g * (acc[t][r] / lrun[r]) + x[off];
    }
  }
}

extern "C" void kernel_launch(void* const* d_in, const int* in_sizes, int n_in,
                              void* d_out, int out_size, void* d_ws, size_t ws_size,
                              hipStream_t stream) {
  (void)in_sizes; (void)n_in; (void)out_size; (void)ws_size;
  const float* x     = (const float*)d_in[0];
  const float* mask  = (const float*)d_in[1];
  const float* Wq    = (const float*)d_in[2];
  const float* bq    = (const float*)d_in[3];
  const float* Wk    = (const float*)d_in[4];
  const float* bk    = (const float*)d_in[5];
  const float* Wv    = (const float*)d_in[6];
  const float* bv    = (const float*)d_in[7];
  const float* gamma = (const float*)d_in[8];
  float* out = (float*)d_out;

  char* ws = (char*)d_ws;
  size_t off = 0;
  auto take = [&](size_t bytes) -> char* {
    char* p = ws + off;
    off += (bytes + 255) & ~(size_t)255;
    return p;
  };
  _Float16* Wh = (_Float16*)take((size_t)320 * 256 * sizeof(_Float16));  // 160 KB
  float*    bP = (float*)take(320 * sizeof(float));
  _Float16* xt = (_Float16*)take((size_t)B_ * N_ * C_ * sizeof(_Float16));   // 8 MB
  _Float16* Qh = (_Float16*)take((size_t)B_ * N_ * CQ_ * sizeof(_Float16)); // 1 MB
  _Float16* Kh = (_Float16*)take((size_t)B_ * N_ * CQ_ * sizeof(_Float16)); // 1 MB
  _Float16* Vh = (_Float16*)take((size_t)B_ * C_ * N_ * sizeof(_Float16));  // 8 MB

  pack_weights<<<(320 * 256 + 255) / 256, 256, 0, stream>>>(Wq, bq, Wk, bk, Wv, bv, Wh, bP);
  transpose_cvt<<<dim3(N_ / 32, C_ / 32, B_), 256, 0, stream>>>(x, xt);
  proj_kernel<<<dim3(N_ / 16, 320 / 16, B_), 32, 0, stream>>>(Wh, bP, xt, mask, Qh, Kh, Vh);
  attn_kernel<<<dim3(N_ / 16, B_), 64, 0, stream>>>(Qh, Kh, Vh, x, gamma, out);
}
